// _LightVLACore_35570919145560
// MI455X (gfx1250) — compile-verified, matrix operand order checked by reading, not run on hardware
//
#include <hip/hip_runtime.h>
#include <hip/hip_bf16.h>

// ---------------------------------------------------------------------------
// Problem constants (B,P,T,D fixed by the reference; keep_tokens = 256)
// ---------------------------------------------------------------------------
static constexpr int B = 8;
static constexpr int P = 1024;
static constexpr int T = 64;
static constexpr int D = 2048;
static constexpr int KEEP = 256;
static constexpr float EPS = 1e-6f;
static constexpr float SCALE = 0.022097086912079608f; // 1/sqrt(2048)

typedef __bf16 bf16_t;
typedef __attribute__((ext_vector_type(16))) __bf16 v16bf;
typedef __attribute__((ext_vector_type(8)))  __bf16 v8bf;
typedef __attribute__((ext_vector_type(8)))  float  v8f;
typedef __attribute__((ext_vector_type(4)))  int    v4i;

// float -> bf16 (round to nearest even)
__device__ __forceinline__ unsigned short f2bf(float f) {
    unsigned int u = __float_as_uint(f);
    u += 0x7FFFu + ((u >> 16) & 1u);
    return (unsigned short)(u >> 16);
}

// Load one lane's A/B fragment half for 16x16x32 bf16 WMMA.
// Per ISA layout: a lane holds K = base+{0..7} (regs 0-3) and base+{16..23}
// (regs 4-7); the caller folds the +8 for lanes 16-31 into the pointer.
__device__ __forceinline__ v16bf load_frag16(const bf16_t* p) {
    v8bf lo = *(const v8bf*)(p);
    v8bf hi = *(const v8bf*)(p + 16);
    return __builtin_shufflevector(lo, hi, 0, 1, 2, 3, 4, 5, 6, 7,
                                   8, 9, 10, 11, 12, 13, 14, 15);
}

__device__ __forceinline__ v8f wmma_bf16(v16bf a, v16bf b, v8f c) {
    return __builtin_amdgcn_wmma_f32_16x16x32_bf16(
        /*neg_a=*/false, a, /*neg_b=*/false, b,
        /*c_mod=*/(short)0, c, /*reuse_a=*/false, /*reuse_b=*/false);
}

// ---------------------------------------------------------------------------
// CDNA5 async global->LDS copy (16 bytes), tracked by ASYNCcnt.
// Builtin signature wants (v4i AS1* src, v4i AS3* dst, imm offset, imm cpol).
// ---------------------------------------------------------------------------
__device__ __forceinline__ void async_copy16(const unsigned short* g,
                                             unsigned short* l) {
#if __has_builtin(__builtin_amdgcn_global_load_async_to_lds_b128)
    __builtin_amdgcn_global_load_async_to_lds_b128(
        (__attribute__((address_space(1))) v4i*)g,
        (__attribute__((address_space(3))) v4i*)l, 0, 0);
#else
    unsigned int loff = (unsigned int)(size_t)l;  // LDS byte offset (addr[31:0])
    asm volatile("global_load_async_to_lds_b128 %0, %1, off"
                 :: "v"(loff), "v"(g) : "memory");
#endif
}

template <int N>
__device__ __forceinline__ void wait_async() {
#if __has_builtin(__builtin_amdgcn_s_wait_asynccnt)
    __builtin_amdgcn_s_wait_asynccnt(N);
#else
    asm volatile("s_wait_asynccnt %0" :: "i"(N) : "memory");
#endif
}

// ---------------------------------------------------------------------------
// Kernel 1: RMS-norm patches [B*P rows of D] fp32 -> bf16
// ---------------------------------------------------------------------------
__global__ void k_rms_patches(const float* __restrict__ x,
                              unsigned short* __restrict__ out) {
    __shared__ float warpsum[8];
    __shared__ float total;
    const int row = blockIdx.x;          // B*P rows
    const int tid = threadIdx.x;         // 256 threads
    const int lane = tid & 31, wave = tid >> 5;
    const float* px = x + (size_t)row * D;
    float ss = 0.f;
    for (int i = tid; i < D; i += 256) { float v = px[i]; ss += v * v; }
    #pragma unroll
    for (int m = 16; m >= 1; m >>= 1) ss += __shfl_xor(ss, m);
    if (lane == 0) warpsum[wave] = ss;
    __syncthreads();
    if (tid == 0) {
        float s = 0.f;
        #pragma unroll
        for (int i = 0; i < 8; i++) s += warpsum[i];
        total = s;
    }
    __syncthreads();
    const float inv = rsqrtf(total / (float)D + EPS);
    unsigned short* po = out + (size_t)row * D;
    for (int i = tid; i < D; i += 256) po[i] = f2bf(px[i] * inv);
}

// ---------------------------------------------------------------------------
// Kernel 2: RMS-norm task tokens; writes row-major [B,T,D] and transposed
// [B,D,T] bf16 copies (transposed copy feeds the attn@task_n GEMM B-operand).
// ---------------------------------------------------------------------------
__global__ void k_rms_task(const float* __restrict__ x,
                           unsigned short* __restrict__ out_n,
                           unsigned short* __restrict__ out_t) {
    __shared__ float warpsum[8];
    __shared__ float total;
    const int row = blockIdx.x;          // B*T rows
    const int b = row / T, t = row % T;
    const int tid = threadIdx.x;
    const int lane = tid & 31, wave = tid >> 5;
    const float* px = x + (size_t)row * D;
    float ss = 0.f;
    for (int i = tid; i < D; i += 256) { float v = px[i]; ss += v * v; }
    #pragma unroll
    for (int m = 16; m >= 1; m >>= 1) ss += __shfl_xor(ss, m);
    if (lane == 0) warpsum[wave] = ss;
    __syncthreads();
    if (tid == 0) {
        float s = 0.f;
        #pragma unroll
        for (int i = 0; i < 8; i++) s += warpsum[i];
        total = s;
    }
    __syncthreads();
    const float inv = rsqrtf(total / (float)D + EPS);
    for (int i = tid; i < D; i += 256) {
        unsigned short v = f2bf(px[i] * inv);
        out_n[(size_t)row * D + i] = v;                       // [B,T,D]
        out_t[((size_t)b * D + i) * T + t] = v;               // [B,D,T]
    }
}

// ---------------------------------------------------------------------------
// Kernel 3: fused attention logits + softmax.
// Wave computes a 16x64 logits tile (all of T) via K=2048 WMMA loop,
// then row-softmax with 16-lane shuffles; writes attn bf16 [B,P,T].
// Grid: B*(P/128) blocks of 256 threads (8 waves x 16 rows).
// ---------------------------------------------------------------------------
__global__ void k_attn_softmax(const unsigned short* __restrict__ pn,
                               const unsigned short* __restrict__ tn,
                               unsigned short* __restrict__ attn) {
    const int lane = threadIdx.x & 31;
    const int wave = threadIdx.x >> 5;
    const int blk = blockIdx.x;
    const int b = blk / (P / 128);
    const int p_base = (blk % (P / 128)) * 128 + wave * 16;
    const int lrow = lane & 15;
    const int khalf = (lane >> 4) << 3;  // +8 K-offset for upper half-wave

    const bf16_t* A = (const bf16_t*)pn + ((size_t)(b * P + p_base + lrow)) * D;
    const bf16_t* Bm = (const bf16_t*)tn + ((size_t)b * T) * D;
    const bf16_t* B0 = Bm + (size_t)(0 * 16 + lrow) * D;
    const bf16_t* B1 = Bm + (size_t)(1 * 16 + lrow) * D;
    const bf16_t* B2 = Bm + (size_t)(2 * 16 + lrow) * D;
    const bf16_t* B3 = Bm + (size_t)(3 * 16 + lrow) * D;

    v8f c0 = {}, c1 = {}, c2 = {}, c3 = {};
    for (int k0 = 0; k0 < D; k0 += 32) {
        const int kk = k0 + khalf;
        v16bf a  = load_frag16(A + kk);
        v16bf b0 = load_frag16(B0 + kk);
        v16bf b1 = load_frag16(B1 + kk);
        v16bf b2 = load_frag16(B2 + kk);
        v16bf b3 = load_frag16(B3 + kk);
        c0 = wmma_bf16(a, b0, c0);
        c1 = wmma_bf16(a, b1, c1);
        c2 = wmma_bf16(a, b2, c2);
        c3 = wmma_bf16(a, b3, c3);
    }
    // C layout: lanes 0-15 hold row r (col=lane), lanes 16-31 hold row r+8.
    #pragma unroll
    for (int r = 0; r < 8; r++) {
        float v0 = c0[r] * SCALE, v1 = c1[r] * SCALE;
        float v2 = c2[r] * SCALE, v3 = c3[r] * SCALE;
        float m = fmaxf(fmaxf(v0, v1), fmaxf(v2, v3));
        #pragma unroll
        for (int s = 8; s >= 1; s >>= 1) m = fmaxf(m, __shfl_xor(m, s));
        float e0 = __expf(v0 - m), e1 = __expf(v1 - m);
        float e2 = __expf(v2 - m), e3 = __expf(v3 - m);
        float sum = e0 + e1 + e2 + e3;
        #pragma unroll
        for (int s = 8; s >= 1; s >>= 1) sum += __shfl_xor(sum, s);
        const float inv = 1.0f / sum;
        const int row = p_base + r + ((lane >> 4) << 3);
        unsigned short* po = attn + ((size_t)(b * P + row)) * T + lrow;
        po[0]  = f2bf(e0 * inv);
        po[16] = f2bf(e1 * inv);
        po[32] = f2bf(e2 * inv);
        po[48] = f2bf(e3 * inv);
    }
}

// ---------------------------------------------------------------------------
// Kernel 4: queries = attn @ task_n   ([B,P,T] x [B,T,D] -> [B,P,D] bf16).
// K=64 -> 2 WMMA steps. B-operand from the transposed task copy [B,D,T] so
// each lane's K-run is contiguous. RMS-norm of queries is SKIPPED: it is a
// positive per-row scale and score feeds a per-row argmax (scale-invariant).
// Grid: B*(P/16)*2 blocks of 256 threads; wave w covers 128 columns.
// ---------------------------------------------------------------------------
__global__ void k_queries(const unsigned short* __restrict__ attn,
                          const unsigned short* __restrict__ tnT,
                          unsigned short* __restrict__ q) {
    const int lane = threadIdx.x & 31;
    const int wave = threadIdx.x >> 5;
    int blk = blockIdx.x;
    const int half = blk & 1; blk >>= 1;
    const int ptile = blk % (P / 16);
    const int b = blk / (P / 16);
    const int lrow = lane & 15;
    const int khalf = (lane >> 4) << 3;
    const int n_base = half * 1024 + wave * 128;

    const bf16_t* A = (const bf16_t*)attn +
                      ((size_t)(b * P + ptile * 16 + lrow)) * T;
    const bf16_t* Bt = (const bf16_t*)tnT + (size_t)b * D * T;

    v8f c[8] = {};
    #pragma unroll
    for (int k0 = 0; k0 < T; k0 += 32) {
        const int kk = k0 + khalf;
        v16bf a = load_frag16(A + kk);
        #pragma unroll
        for (int t = 0; t < 8; t++) {
            const int n = n_base + t * 16 + lrow;
            v16bf bb = load_frag16(Bt + (size_t)n * T + kk);
            c[t] = wmma_bf16(a, bb, c[t]);
        }
    }
    #pragma unroll
    for (int t = 0; t < 8; t++) {
        #pragma unroll
        for (int r = 0; r < 8; r++) {
            const int row = ptile * 16 + r + ((lane >> 4) << 3);
            const int col = n_base + t * 16 + lrow;
            q[((size_t)(b * P + row)) * D + col] = f2bf(c[t][r]);
        }
    }
}

// ---------------------------------------------------------------------------
// Kernel 5: fused score GEMM + running per-row argmax.
// Workgroup = 256 threads (8 waves) owns 128 score rows. For each 128-column
// chunk of P, the K=2048 loop is staged through LDS in 64-wide K-chunks with
// DOUBLE-BUFFERED async global->LDS copies (ASYNCcnt): issue chunk i+1,
// s_wait_asynccnt<=8 (in-order completion => chunk i landed), barrier,
// WMMA out of LDS (ds_load_b128, conflict-free via 72-element row padding).
// B-tiles are shared by all 8 waves (8x global-traffic reduction vs r1).
// Dynamic LDS: 2 bufs x (A 128x72 + B 128x72) bf16 = 73,728 bytes.
// Grid: B*(P/128) x 256.
// ---------------------------------------------------------------------------
static constexpr int LDS_ROW = 72;                 // 64 K-elems + 8 pad (144B)
static constexpr int A_OFF   = 2 * 128 * LDS_ROW;  // elements
static constexpr int SMEM_BYTES = 2 * 2 * 128 * LDS_ROW * 2; // 73,728

__global__ void k_score_argmax(const unsigned short* __restrict__ q,
                               const unsigned short* __restrict__ pn,
                               int* __restrict__ indices) {
    extern __shared__ unsigned short smem[];  // [A: 2][128][72] [B: 2][128][72]
    const int tid = threadIdx.x;
    const int lane = tid & 31;
    const int wave = tid >> 5;
    const int blk = blockIdx.x;
    const int b = blk / (P / 128);
    const int R0 = (blk % (P / 128)) * 128;   // first score row of this WG
    const int p_base = R0 + wave * 16;
    const int lrow = lane & 15;
    const int khalf = (lane >> 4) << 3;

    const unsigned short* Ag = q + ((size_t)(b * P + R0)) * D;   // 128 A rows
    const unsigned short* Pg = pn + (size_t)b * P * D;

    float bestv[8];
    int   besti[8];
    #pragma unroll
    for (int r = 0; r < 8; r++) { bestv[r] = -3.4e38f; besti[r] = 0; }

    // per-thread share of one 64-K chunk: 4 A + 4 B 16-byte transfers
    auto stage = [&](const unsigned short* Bg, int buf, int k0) {
        #pragma unroll
        for (int it = 0; it < 4; ++it) {
            const int idx = it * 256 + tid;        // 0..1023
            const int row = idx >> 3;              // 0..127
            const int seg = (idx & 7) * 8;         // K-elem offset within chunk
            async_copy16(Ag + (size_t)row * D + k0 + seg,
                         smem + (buf * 128 + row) * LDS_ROW + seg);
            async_copy16(Bg + (size_t)row * D + k0 + seg,
                         smem + A_OFF + (buf * 128 + row) * LDS_ROW + seg);
        }
    };

    for (int q0 = 0; q0 < P; q0 += 128) {
        const unsigned short* Bg = Pg + (size_t)q0 * D;  // 128 B rows (cols)
        v8f c[8] = {};
        stage(Bg, 0, 0);
        for (int ch = 0; ch < D / 64; ++ch) {
            const int buf = ch & 1;
            if (ch + 1 < D / 64) {
                stage(Bg, buf ^ 1, (ch + 1) * 64);
                wait_async<8>();   // chunk `ch` complete (in-order)
            } else {
                wait_async<0>();
            }
            __syncthreads();       // all waves' copies visible
            const unsigned short* As =
                smem + (buf * 128 + wave * 16 + lrow) * LDS_ROW + khalf;
            const unsigned short* Bs =
                smem + A_OFF + (buf * 128 + lrow) * LDS_ROW + khalf;
            #pragma unroll
            for (int kcs = 0; kcs < 2; ++kcs) {
                const int kc = kcs * 32;
                v16bf a = load_frag16((const bf16_t*)(As + kc));
                #pragma unroll
                for (int t = 0; t < 8; ++t) {
                    v16bf bb = load_frag16(
                        (const bf16_t*)(Bs + t * 16 * LDS_ROW + kc));
                    c[t] = wmma_bf16(a, bb, c[t]);
                }
            }
            __syncthreads();       // done reading buf before refill at ch+2
        }
        #pragma unroll
        for (int t = 0; t < 8; ++t) {
            const int col = q0 + t * 16 + lrow;
            #pragma unroll
            for (int r = 0; r < 8; ++r) {
                const float v = c[t][r];
                if (v > bestv[r]) { bestv[r] = v; besti[r] = col; }
            }
        }
    }
    #pragma unroll
    for (int r = 0; r < 8; r++) {
        #pragma unroll
        for (int m = 8; m >= 1; m >>= 1) {
            float ov = __shfl_xor(bestv[r], m);
            int   oi = __shfl_xor(besti[r], m);
            if (ov > bestv[r] || (ov == bestv[r] && oi < besti[r])) {
                bestv[r] = ov; besti[r] = oi;
            }
        }
        if (lrow == 0) {
            const int row = p_base + r + ((lane >> 4) << 3);
            indices[b * P + row] = besti[r];
        }
    }
}

// ---------------------------------------------------------------------------
// Kernel 6/7: zero counts; histogram of argmax indices.
// ---------------------------------------------------------------------------
__global__ void k_zero_counts(int* __restrict__ counts) {
    const int i = blockIdx.x * 256 + threadIdx.x;
    if (i < B * P) counts[i] = 0;
}

__global__ void k_histogram(const int* __restrict__ indices,
                            int* __restrict__ counts) {
    const int i = blockIdx.x * 256 + threadIdx.x;
    if (i < B * P) {
        const int b = i / P;
        atomicAdd(&counts[b * P + indices[i]], 1);
    }
}

// ---------------------------------------------------------------------------
// Kernel 8: exact top-k by rank (matches jax.lax.top_k tie-break: higher
// count first, lower index among equals). Ranks form a total order so
// exactly KEEP rows are selected; LDS prefix-sum compacts them in
// ascending index order == jnp.sort(topk_idx). One 1024-thread WG / batch.
// ---------------------------------------------------------------------------
__global__ void k_topk_select(const int* __restrict__ counts,
                              int* __restrict__ gidx) {
    __shared__ int c[P];
    __shared__ int s[P];
    const int b = blockIdx.x;
    const int j = threadIdx.x;
    c[j] = counts[b * P + j];
    __syncthreads();
    const int cj = c[j];
    int rank = 0;
    for (int i = 0; i < P; i++) {
        const int ci = c[i];
        rank += (ci > cj) || (ci == cj && i < j);
    }
    const int sel = (rank < KEEP) ? 1 : 0;
    s[j] = sel;
    __syncthreads();
    for (int off = 1; off < P; off <<= 1) {      // Hillis-Steele inclusive scan
        const int t = (j >= off) ? s[j - off] : 0;
        __syncthreads();
        s[j] += t;
        __syncthreads();
    }
    if (sel) gidx[b * KEEP + (s[j] - 1)] = j;
}

// ---------------------------------------------------------------------------
// Kernel 9: gather kept rows of ORIGINAL fp32 patches; write mask = 1.
// ---------------------------------------------------------------------------
__global__ void k_gather(const float* __restrict__ patches,
                         const int* __restrict__ gidx,
                         float* __restrict__ out) {
    const int blk = blockIdx.x;          // B*KEEP
    const int b = blk / KEEP;
    const int src_row = gidx[blk];
    const float* src = patches + ((size_t)(b * P + src_row)) * D;
    float* dst = out + (size_t)blk * D;
    for (int i = threadIdx.x; i < D; i += 256) dst[i] = src[i];
    if (threadIdx.x == 0) {
        float* mask = out + (size_t)B * KEEP * D;
        mask[blk] = 1.0f;
    }
}

// ---------------------------------------------------------------------------
// Launch
// ---------------------------------------------------------------------------
extern "C" void kernel_launch(void* const* d_in, const int* in_sizes, int n_in,
                              void* d_out, int out_size, void* d_ws, size_t ws_size,
                              hipStream_t stream) {
    (void)in_sizes; (void)n_in; (void)out_size; (void)ws_size;
    const float* patches = (const float*)d_in[0];
    const float* task    = (const float*)d_in[1];
    float* out = (float*)d_out;

    char* ws = (char*)d_ws;
    size_t off = 0;
    auto alloc = [&](size_t bytes) {
        void* p = ws + off;
        off += (bytes + 255) & ~(size_t)255;
        return p;
    };
    unsigned short* pn   = (unsigned short*)alloc((size_t)B * P * D * 2); // patches_n bf16
    unsigned short* tn   = (unsigned short*)alloc((size_t)B * T * D * 2); // task_n bf16 [B,T,D]
    unsigned short* tnT  = (unsigned short*)alloc((size_t)B * D * T * 2); // task_n bf16 [B,D,T]
    unsigned short* attn = (unsigned short*)alloc((size_t)B * P * T * 2); // attn bf16
    unsigned short* qbuf = (unsigned short*)alloc((size_t)B * P * D * 2); // queries bf16
    int* indices = (int*)alloc((size_t)B * P * 4);
    int* counts  = (int*)alloc((size_t)B * P * 4);
    int* gidx    = (int*)alloc((size_t)B * KEEP * 4);

    k_rms_patches<<<B * P, 256, 0, stream>>>(patches, pn);
    k_rms_task<<<B * T, 256, 0, stream>>>(task, tn, tnT);
    k_attn_softmax<<<B * (P / 128), 256, 0, stream>>>(pn, tn, attn);
    k_queries<<<B * (P / 16) * 2, 256, 0, stream>>>(attn, tnT, qbuf);
    k_score_argmax<<<B * (P / 128), 256, SMEM_BYTES, stream>>>(qbuf, pn, indices);
    k_zero_counts<<<(B * P) / 256, 256, 0, stream>>>(counts);
    k_histogram<<<(B * P) / 256, 256, 0, stream>>>(indices, counts);
    k_topk_select<<<B, 1024, 0, stream>>>(counts, gidx);
    k_gather<<<B * KEEP, 256, 0, stream>>>(patches, gidx, out);
}
